// PlanStructuredNetwork_515396076298
// MI455X (gfx1250) — matrix operand order repaired and checked by reading, not compiled, not run on hardware
//
#include <hip/hip_runtime.h>
#include <hip/hip_bf16.h>

// ---------------------------------------------------------------------------
// PlanStructuredNetwork on MI455X (gfx1250)
// Compute-bound tree-of-MLPs: bf16 WMMA (16x16x32, f32 accum), activations
// LDS-resident across the 4 MLP layers, weights packed once into the WMMA
// B-fragment layout so each lane fetches its fragment with 2 b128 loads.
// ---------------------------------------------------------------------------

typedef __bf16 bf16_t;
typedef __bf16 v16bf __attribute__((ext_vector_type(16)));
typedef __bf16 v8bf  __attribute__((ext_vector_type(8)));
typedef float  v8f   __attribute__((ext_vector_type(8)));

#define B_      256
#define F_      64
#define H_      512
#define OUT_    65
#define OUTP_   80      // 65 padded to 5 col-tiles of 16
#define NLEAF_  512
#define NINT_   511
#define NNODE_  1023
#define IIN_    194     // 64 + 65 + 65
#define IINP_   224     // padded to multiple of 32
#define ROWS_   64      // rows per workgroup
#define THREADS_ 256    // 8 waves (wave32)

// LDS budgets (dynamic shared memory; WGP has 320KB)
#define LEAF_SMEM (ROWS_*F_*2   + 2*ROWS_*H_*2)   // 8K  + 128K = 139264
#define INT_SMEM  (ROWS_*IINP_*2 + 2*ROWS_*H_*2)  // 28K + 128K = 159744

// ---------------------------------------------------------------------------
// Fragment loaders (contiguous k order k = 16*half + e, same for A and B)
// ---------------------------------------------------------------------------

// A: 16x32 bf16 tile from LDS, row-major with row stride ldk (elements).
// lanes 0-15 row M=lane: VGPR0-7 = K 0..15
// lanes 16-31 row M=lane-16: VGPR0-7 = K 16..31
__device__ __forceinline__ v16bf load_a_frag(const bf16_t* src, int ldk,
                                             int mt, int kt, int lane) {
    const int half = lane >> 4, r = lane & 15;
    const bf16_t* p = src + (mt * 16 + r) * ldk + kt * 32 + half * 16;
    v8bf a0 = *(const v8bf*)(p);        // K base..base+7   -> VGPR0..3
    v8bf a1 = *(const v8bf*)(p + 8);    // K base+8..+15    -> VGPR4..7
    return __builtin_shufflevector(a0, a1, 0,1,2,3,4,5,6,7,8,9,10,11,12,13,14,15);
}

// B: 32x16 bf16 tile from global, pre-packed so each lane's 16 values are
// contiguous (32 bytes): packed[((kt*NT+nt)*32 + lane)*16 + e]
__device__ __forceinline__ v16bf load_b_frag(const bf16_t* __restrict__ wp,
                                             int NT, int kt, int nt, int lane) {
    const bf16_t* p = wp + (size_t)((kt * NT + nt) * 32 + lane) * 16;
    v8bf b0 = *(const v8bf*)(p);
    v8bf b1 = *(const v8bf*)(p + 8);
    return __builtin_shufflevector(b0, b1, 0,1,2,3,4,5,6,7,8,9,10,11,12,13,14,15);
}

// ---------------------------------------------------------------------------
// One hidden layer: dst = relu(src @ W + b), src 64xK (LDS, bf16),
// W packed KxH_ (global bf16), dst 64xH_ (LDS, bf16).
// Wave w: row-tile mt = w&3; col-tiles nt = (w>>2) + 2t, t=0..15.
// ---------------------------------------------------------------------------
__device__ __forceinline__ void layer_to_lds(const bf16_t* src, int ldk, int ksteps,
                                             const bf16_t* __restrict__ wp,
                                             const float* __restrict__ bias,
                                             bf16_t* dst) {
    const int tid = threadIdx.x, wave = tid >> 5, lane = tid & 31;
    const int mt = wave & 3;
    const int ntBase = wave >> 2;          // 0 or 1
    v8f acc[16] = {};
    for (int kt = 0; kt < ksteps; ++kt) {
        v16bf a = load_a_frag(src, ldk, mt, kt, lane);
#pragma unroll
        for (int t = 0; t < 16; ++t) {
            const int nt = ntBase + 2 * t;
            v16bf b = load_b_frag(wp, H_ / 16, kt, nt, lane);
            acc[t] = __builtin_amdgcn_wmma_f32_16x16x32_bf16(
                false, a, false, b, (short)0, acc[t], false, false);
        }
    }
    // epilogue: C/D layout -> lane holds col n = lane&15, rows r + 8*(lane>>4)
    const int half = lane >> 4, n0 = lane & 15;
#pragma unroll
    for (int t = 0; t < 16; ++t) {
        const int nt = ntBase + 2 * t;
        const int n = nt * 16 + n0;
        const float bv = bias[n];
#pragma unroll
        for (int r = 0; r < 8; ++r) {
            float v = acc[t][r] + bv;
            v = v > 0.f ? v : 0.f;
            dst[(mt * 16 + half * 8 + r) * H_ + n] = (bf16_t)v;
        }
    }
}

// ---------------------------------------------------------------------------
// Output layer: nodeout = src @ Wo + bo  (no relu), N padded 65->80 (NT=5).
// mode 0: leaf rows (grow = b*512 + leaf, node = 511 + leaf)
// mode 1: internal rows (grow = b*2^shift + j, node = nodeBase + j)
// ---------------------------------------------------------------------------
__device__ __forceinline__ void layer_to_global(const bf16_t* src,
                                                const bf16_t* __restrict__ wp,
                                                const float* __restrict__ bias,
                                                float* __restrict__ nodeout,
                                                int grow0, int mode,
                                                int cntShift, int nodeBase) {
    const int tid = threadIdx.x, wave = tid >> 5, lane = tid & 31;
    const int mt = wave & 3, g = wave >> 2;
    int nts[3];
    int ntn;
    if (g == 0) { nts[0] = 0; nts[1] = 2; nts[2] = 4; ntn = 3; }
    else        { nts[0] = 1; nts[1] = 3; nts[2] = 3; ntn = 2; }
    v8f acc[3] = {};
    for (int kt = 0; kt < H_ / 32; ++kt) {
        v16bf a = load_a_frag(src, H_, mt, kt, lane);
#pragma unroll
        for (int t = 0; t < 3; ++t) {
            if (t < ntn) {
                v16bf b = load_b_frag(wp, OUTP_ / 16, kt, nts[t], lane);
                acc[t] = __builtin_amdgcn_wmma_f32_16x16x32_bf16(
                    false, a, false, b, (short)0, acc[t], false, false);
            }
        }
    }
    const int half = lane >> 4, n0 = lane & 15;
#pragma unroll
    for (int t = 0; t < 3; ++t) {
        if (t >= ntn) continue;
        const int n = nts[t] * 16 + n0;
        if (n >= OUT_) continue;
        const float bv = bias[n];
#pragma unroll
        for (int r = 0; r < 8; ++r) {
            const int grow = grow0 + mt * 16 + half * 8 + r;
            int bIdx, node;
            if (mode == 0) { bIdx = grow >> 9;        node = NINT_ + (grow & (NLEAF_ - 1)); }
            else           { bIdx = grow >> cntShift; node = nodeBase + (grow & ((1 << cntShift) - 1)); }
            nodeout[((size_t)bIdx * NNODE_ + node) * OUT_ + n] = acc[t][r] + bv;
        }
    }
}

// ---------------------------------------------------------------------------
// Kernels
// ---------------------------------------------------------------------------

extern "C" __global__ __launch_bounds__(THREADS_)
void leaf_mlp_kernel(const float* __restrict__ leaf_feats,
                     const bf16_t* __restrict__ w0p, const float* __restrict__ b0,
                     const bf16_t* __restrict__ w1p, const float* __restrict__ b1,
                     const bf16_t* __restrict__ w2p, const float* __restrict__ b2,
                     const bf16_t* __restrict__ wop, const float* __restrict__ bo,
                     float* __restrict__ nodeout) {
    extern __shared__ char smem[];
    bf16_t* X  = (bf16_t*)smem;                        // 64 x 64
    bf16_t* Ha = (bf16_t*)(smem + ROWS_ * F_ * 2);     // 64 x 512
    bf16_t* Hb = Ha + ROWS_ * H_;                      // 64 x 512
    const int grow0 = blockIdx.x * ROWS_;
    // stage X (rows are contiguous in leaf_feats: (B*N_LEAF, F) row-major)
    for (int idx = threadIdx.x; idx < ROWS_ * F_; idx += THREADS_)
        X[idx] = (bf16_t)leaf_feats[(size_t)grow0 * F_ + idx];
    __syncthreads();
    layer_to_lds(X, F_, F_ / 32, w0p, b0, Ha);
    __syncthreads();
    layer_to_lds(Ha, H_, H_ / 32, w1p, b1, Hb);
    __syncthreads();
    layer_to_lds(Hb, H_, H_ / 32, w2p, b2, Ha);
    __syncthreads();
    layer_to_global(Ha, wop, bo, nodeout, grow0, 0, 0, 0);
}

extern "C" __global__ __launch_bounds__(THREADS_)
void internal_mlp_kernel(const float* __restrict__ internal_feats,
                         const bf16_t* __restrict__ w0p, const float* __restrict__ b0,
                         const bf16_t* __restrict__ w1p, const float* __restrict__ b1,
                         const bf16_t* __restrict__ w2p, const float* __restrict__ b2,
                         const bf16_t* __restrict__ wop, const float* __restrict__ bo,
                         float* __restrict__ nodeout,
                         int s, int cntShift) {
    extern __shared__ char smem[];
    bf16_t* X  = (bf16_t*)smem;                        // 64 x 224
    bf16_t* Ha = (bf16_t*)(smem + ROWS_ * IINP_ * 2);  // 64 x 512
    bf16_t* Hb = Ha + ROWS_ * H_;
    const int grow0 = blockIdx.x * ROWS_;
    const int cnt = 1 << cntShift;
    // gather concat(internal_feats, left child out, right child out) -> bf16 LDS
    for (int idx = threadIdx.x; idx < ROWS_ * IINP_; idx += THREADS_) {
        const int r = idx / IINP_;
        const int c = idx - r * IINP_;
        const int grow = grow0 + r;
        const int bIdx = grow >> cntShift;
        const int j = grow & (cnt - 1);
        const int node = s + j;
        float v = 0.f;
        if (c < F_)
            v = internal_feats[((size_t)bIdx * NINT_ + node) * F_ + c];
        else if (c < F_ + OUT_)
            v = nodeout[((size_t)bIdx * NNODE_ + 2 * node + 1) * OUT_ + (c - F_)];
        else if (c < F_ + 2 * OUT_)
            v = nodeout[((size_t)bIdx * NNODE_ + 2 * node + 2) * OUT_ + (c - F_ - OUT_)];
        X[idx] = (bf16_t)v;
    }
    __syncthreads();
    layer_to_lds(X, IINP_, IINP_ / 32, w0p, b0, Ha);
    __syncthreads();
    layer_to_lds(Ha, H_, H_ / 32, w1p, b1, Hb);
    __syncthreads();
    layer_to_lds(Hb, H_, H_ / 32, w2p, b2, Ha);
    __syncthreads();
    layer_to_global(Ha, wop, bo, nodeout, grow0, 1, cntShift, s);
}

// Pack fp32 KxN (row-major, real dims Kreal x Nreal) into bf16 WMMA B-tile
// layout with padded dims Kp x Np: packed[((kt*NT+nt)*32+lane)*16 + 2i+j] =
// W[kt*32 + (lane>>4)*16 + 2i + j][nt*16 + (lane&15)]
extern "C" __global__ void pack_weight_kernel(const float* __restrict__ W,
                                              bf16_t* __restrict__ out,
                                              int Kp, int Np, int Kreal, int Nreal) {
    const int total = Kp * Np;
    const int NT = Np / 16;
    for (int t = blockIdx.x * blockDim.x + threadIdx.x; t < total;
         t += gridDim.x * blockDim.x) {
        const int e = t & 15;
        const int lane = (t >> 4) & 31;
        const int tile = t >> 9;
        const int nt = tile % NT;
        const int kt = tile / NT;
        const int n = nt * 16 + (lane & 15);
        const int k = kt * 32 + (lane >> 4) * 16 + e;
        const float v = (k < Kreal && n < Nreal) ? W[(size_t)k * Nreal + n] : 0.f;
        out[t] = (bf16_t)v;
    }
}

extern "C" __global__ void extract_kernel(const float* __restrict__ nodeout,
                                          float* __restrict__ out) {
    const int b = blockIdx.x * blockDim.x + threadIdx.x;
    if (b < B_) out[b] = nodeout[(size_t)b * NNODE_ * OUT_];  // node 0, col 0
}

// ---------------------------------------------------------------------------
// Host driver
// ---------------------------------------------------------------------------
static inline size_t align256(size_t x) { return (x + 255) & ~(size_t)255; }

extern "C" void kernel_launch(void* const* d_in, const int* in_sizes, int n_in,
                              void* d_out, int out_size, void* d_ws, size_t ws_size,
                              hipStream_t stream) {
    (void)in_sizes; (void)n_in; (void)out_size; (void)ws_size;
    const float* leaf_feats     = (const float*)d_in[0];
    const float* internal_feats = (const float*)d_in[1];
    const float* lw0 = (const float*)d_in[2];  const float* lb0 = (const float*)d_in[3];
    const float* lw1 = (const float*)d_in[4];  const float* lb1 = (const float*)d_in[5];
    const float* lw2 = (const float*)d_in[6];  const float* lb2 = (const float*)d_in[7];
    const float* lwo = (const float*)d_in[8];  const float* lbo = (const float*)d_in[9];
    const float* iw0 = (const float*)d_in[10]; const float* ib0 = (const float*)d_in[11];
    const float* iw1 = (const float*)d_in[12]; const float* ib1 = (const float*)d_in[13];
    const float* iw2 = (const float*)d_in[14]; const float* ib2 = (const float*)d_in[15];
    const float* iwo = (const float*)d_in[16]; const float* ibo = (const float*)d_in[17];

    char* ws = (char*)d_ws;
    size_t off = 0;
    float* nodeout = (float*)(ws + off);
    off += align256((size_t)B_ * NNODE_ * OUT_ * sizeof(float));
    bf16_t* lw0p = (bf16_t*)(ws + off); off += align256((size_t)F_   * H_    * 2);
    bf16_t* lw1p = (bf16_t*)(ws + off); off += align256((size_t)H_   * H_    * 2);
    bf16_t* lw2p = (bf16_t*)(ws + off); off += align256((size_t)H_   * H_    * 2);
    bf16_t* lwop = (bf16_t*)(ws + off); off += align256((size_t)H_   * OUTP_ * 2);
    bf16_t* iw0p = (bf16_t*)(ws + off); off += align256((size_t)IINP_* H_    * 2);
    bf16_t* iw1p = (bf16_t*)(ws + off); off += align256((size_t)H_   * H_    * 2);
    bf16_t* iw2p = (bf16_t*)(ws + off); off += align256((size_t)H_   * H_    * 2);
    bf16_t* iwop = (bf16_t*)(ws + off); off += align256((size_t)H_   * OUTP_ * 2);

    // one-time (per call) weight conversion into WMMA fragment layout
    pack_weight_kernel<<<256, 256, 0, stream>>>(lw0, lw0p, F_,    H_,    F_,   H_);
    pack_weight_kernel<<<512, 256, 0, stream>>>(lw1, lw1p, H_,    H_,    H_,   H_);
    pack_weight_kernel<<<512, 256, 0, stream>>>(lw2, lw2p, H_,    H_,    H_,   H_);
    pack_weight_kernel<<<256, 256, 0, stream>>>(lwo, lwop, H_,    OUTP_, H_,   OUT_);
    pack_weight_kernel<<<256, 256, 0, stream>>>(iw0, iw0p, IINP_, H_,    IIN_, H_);
    pack_weight_kernel<<<512, 256, 0, stream>>>(iw1, iw1p, H_,    H_,    H_,   H_);
    pack_weight_kernel<<<512, 256, 0, stream>>>(iw2, iw2p, H_,    H_,    H_,   H_);
    pack_weight_kernel<<<256, 256, 0, stream>>>(iwo, iwop, H_,    OUTP_, H_,   OUT_);

    // allow 160KB dynamic LDS (WGP has 320KB)
    (void)hipFuncSetAttribute((const void*)leaf_mlp_kernel,
                              hipFuncAttributeMaxDynamicSharedMemorySize, INT_SMEM);
    (void)hipFuncSetAttribute((const void*)internal_mlp_kernel,
                              hipFuncAttributeMaxDynamicSharedMemorySize, INT_SMEM);

    // leaf stage: 131072 rows -> 2048 workgroups
    leaf_mlp_kernel<<<(B_ * NLEAF_) / ROWS_, THREADS_, LEAF_SMEM, stream>>>(
        leaf_feats, lw0p, lb0, lw1p, lb1, lw2p, lb2, lwop, lbo, nodeout);

    // internal levels, bottom-up (sequential dependency)
    for (int l = 8; l >= 0; --l) {
        const int s = (1 << l) - 1;
        const int wgs = (B_ << l) / ROWS_;   // B * 2^l rows
        internal_mlp_kernel<<<wgs, THREADS_, INT_SMEM, stream>>>(
            internal_feats, iw0p, ib0, iw1p, ib1, iw2p, ib2, iwop, ibo,
            nodeout, s, l);
    }

    extract_kernel<<<1, THREADS_, 0, stream>>>(nodeout, (float*)d_out);
}
